// FieldFormerMLPSparsePollution_67748814127613
// MI455X (gfx1250) — matrix operand order repaired and checked by reading, not compiled, not run on hardware
//
#include <hip/hip_runtime.h>

typedef _Float16 h16;
typedef __attribute__((ext_vector_type(16))) _Float16 v16h;
typedef __attribute__((ext_vector_type(4)))  float    f32x4;
typedef __attribute__((ext_vector_type(8)))  float    v8f;

#define DQ   4            // queries per block
#define KNB  32           // neighbors per query
#define MTOK (DQ * KNB)   // 128 token rows per block
#define DD   128          // hidden dim D
#define FF   256          // hidden dim F
#define ST_TOK 40         // 32 + 8 halves  (80B  = 5*16, lane stride 20 dw -> conflict-free)
#define ST_H   136        // 128 + 8 halves (272B, lane stride 68 dw == 4 mod 64)
#define ST_X   264        // 256 + 8 halves (528B, lane stride 132 dw == 4 mod 64)

// ---- branchless erf-GELU (A&S 7.1.26, |err| < 1.5e-7) ---------------------
// Uses raw v_rcp_f32 / v_exp_f32 (TRANS ops co-issue with WMMA); no div expansion.
__device__ __forceinline__ float gelu_erf(float x) {
    const float z  = x * 0.70710678118654752f;
    const float az = fabsf(z);
    const float t  = __builtin_amdgcn_rcpf(1.0f + 0.3275911f * az);   // v_rcp_f32
    float y = 1.061405429f;
    y = y * t - 1.453152027f;
    y = y * t + 1.421413741f;
    y = y * t - 0.284496736f;
    y = y * t + 0.254829592f;
    const float e   = __expf(-az * az);                               // v_exp_f32
    const float erf = copysignf(1.0f - y * t * e, z);
    return 0.5f * x * (1.0f + erf);
}

// ---- async global -> LDS staging (CDNA5 ASYNCcnt path) --------------------
__device__ __forceinline__ void async_b128_to_lds(const h16* g, h16* l) {
    const uint32_t lds_off = (uint32_t)(uintptr_t)l;   // low 32b of flat = LDS addr
    const uint64_t gaddr   = (uint64_t)(uintptr_t)g;
    asm volatile("global_load_async_to_lds_b128 %0, %1, off"
                 :: "v"(lds_off), "v"(gaddr) : "memory");
}

// stage row-major [N][K] f16 matrix into LDS with padded stride PS (halves)
template <int N, int K, int PS>
__device__ __forceinline__ void stage_weights(const h16* __restrict__ g, h16* s) {
    constexpr int CPR    = K >> 3;      // 16B chunks per row
    constexpr int CHUNKS = N * CPR;
    for (int c = threadIdx.x; c < CHUNKS; c += 256) {
        const int n  = c / CPR;
        const int kc = (c - n * CPR) << 3;
        async_b128_to_lds(g + n * K + kc, s + n * PS + kc);
    }
}

// ---- WMMA fragment helpers (wave32, v_wmma_f32_16x16x32_f16) --------------
// A 16x32 f16: lanes 0-15 row M=lane, halves K[0..7],K[16..23];
//              lanes 16-31 row M=lane-16, halves K[8..15],K[24..31]
__device__ __forceinline__ v16h ld_frag_a_lds(const h16* base, int stride_h,
                                              int row_base, int kbase) {
    const int lane = threadIdx.x & 31;
    const h16* p = base + (row_base + (lane & 15)) * stride_h
                        + kbase + ((lane >> 4) << 3);
    union { f32x4 f[2]; v16h v; } u;
    u.f[0] = *(const f32x4*)(p);        // ds_load_b128
    u.f[1] = *(const f32x4*)(p + 16);   // ds_load_b128
    return u.v;
}

// B 32x16 f16: lane = column N; lanes 0-15 hold K[0..7],K[16..23],
// lanes 16-31 hold K[8..15],K[24..31]. LDS copy is row-major [N][K] with padded stride.
__device__ __forceinline__ v16h ld_frag_b_lds(const h16* W, int pstride,
                                              int ncol_base, int kchunk) {
    const int lane = threadIdx.x & 31;
    const h16* p = W + (ncol_base + (lane & 15)) * pstride
                     + kchunk + ((lane >> 4) << 3);
    union { f32x4 f[2]; v16h v; } u;
    u.f[0] = *(const f32x4*)(p);        // ds_load_b128
    u.f[1] = *(const f32x4*)(p + 16);   // ds_load_b128
    return u.v;
}

__device__ __forceinline__ v8f wmma_f16(v16h a, v16h b, v8f c) {
    return __builtin_amdgcn_wmma_f32_16x16x32_f16(
        false, a, false, b, (short)0, c, false, false);
}

// C/D layout: VGPR j, lane l -> row = j + (l>=16 ? 8 : 0), col = l & 15
__device__ __forceinline__ void store_tile_gelu(h16* dst, int stride_h,
                                                int row_base, int col_base,
                                                v8f c, const float* bias) {
    const int lane = threadIdx.x & 31;
    const int col  = col_base + (lane & 15);
    const int r0   = row_base + ((lane >> 4) << 3);
    const float b  = bias[col];
#pragma unroll
    for (int j = 0; j < 8; ++j)
        dst[(r0 + j) * stride_h + col] = (h16)gelu_erf(c[j] + b);
}

// ---- prep: f32 weights -> f16 (w_in padded K 4->32) -----------------------
__global__ void ff_prep_weights(const float* __restrict__ w_in,
                                const float* __restrict__ w1,
                                const float* __restrict__ w2,
                                h16* __restrict__ w_in_h,
                                h16* __restrict__ w1_h,
                                h16* __restrict__ w2_h) {
    int i = blockIdx.x * blockDim.x + threadIdx.x;
    if (i < DD * 32) {
        int o = i >> 5, k = i & 31;
        w_in_h[i] = (h16)((k < 4) ? w_in[o * 4 + k] : 0.0f);
    }
    int j = i - DD * 32;
    if (j >= 0 && j < FF * DD) w1_h[j] = (h16)w1[j];
    int l = i - DD * 32 - FF * DD;
    if (l >= 0 && l < DD * FF) w2_h[l] = (h16)w2[l];
}

// ---- main fused kernel: 4 queries (128 tokens) per 256-thread block -------
__global__ __launch_bounds__(256)
void ff_main(const float* __restrict__ xyt_q,
             const float* __restrict__ obs_coords,
             const float* __restrict__ obs_vals,
             const int*   __restrict__ nb_idx,
             const float* __restrict__ log_gammas,
             const float* __restrict__ b_in,
             const float* __restrict__ ln1_g, const float* __restrict__ ln1_b,
             const float* __restrict__ b1,    const float* __restrict__ b2,
             const float* __restrict__ hln_g, const float* __restrict__ hln_b,
             const float* __restrict__ hw1,   const float* __restrict__ hb1,
             const float* __restrict__ hw2,   const float* __restrict__ hb2,
             const h16* __restrict__ w_in_h,
             const h16* __restrict__ w1_h,
             const h16* __restrict__ w2_h,
             float* __restrict__ out) {
    extern __shared__ char smem[];
    h16*   sWin  = (h16*)smem;                        // [128][40]
    h16*   sW1   = sWin + DD * ST_TOK;                // [256][136]
    h16*   sW2   = sW1  + FF * ST_H;                  // [128][264]
    h16*   sTok  = sW2  + DD * ST_X;                  // [128][40]
    h16*   sH    = sTok + MTOK * ST_TOK;              // [128][136]
    h16*   sX    = sH   + MTOK * ST_H;                // [128][264]
    float* sV    = (float*)(sX + MTOK * ST_X);        // [4][32]
    float* sMu   = sV   + DQ * KNB;                   // [4]
    float* sSig  = sMu  + DQ;                         // [4]
    float* sPool = sSig + DQ;                         // [4][256] mean|max
    float* sPln  = sPool + DQ * 256;                  // [4][256]
    float* sPv   = sPln  + DQ * 256;                  // [4][256]

    const int tid  = threadIdx.x;
    const int lane = tid & 31;
    const int wv   = tid >> 5;
    const int qblk = blockIdx.x * DQ;

    // ---- kick off async weight staging; overlaps with gather below --------
    stage_weights<DD, 32,  ST_TOK>(w_in_h, sWin);
    stage_weights<FF, DD,  ST_H  >(w1_h,   sW1);
    stage_weights<DD, FF,  ST_X  >(w2_h,   sW2);

    // ---- phase 1: gather neighbors, build tokens --------------------------
    for (int i = tid; i < MTOK * ST_TOK; i += 256) sTok[i] = (h16)0.0f;

    float cx = 0, cy = 0, cz = 0, val = 0, qx = 0, qy = 0, qz = 0;
    if (tid < MTOK) {
        const int q  = tid >> 5, k = tid & 31;
        const int gq = qblk + q;
        const int idx = nb_idx[gq * KNB + k];
        cx = obs_coords[idx * 3 + 0];
        cy = obs_coords[idx * 3 + 1];
        cz = obs_coords[idx * 3 + 2];
        val = obs_vals[idx];
        qx = xyt_q[gq * 3 + 0];
        qy = xyt_q[gq * 3 + 1];
        qz = xyt_q[gq * 3 + 2];
        sV[tid] = val;
    }
    __syncthreads();

    if (tid < DQ) {
        float s = 0.f;
        for (int k = 0; k < KNB; ++k) s += sV[tid * KNB + k];
        const float mu = s * (1.0f / KNB);
        float ss = 0.f;
        for (int k = 0; k < KNB; ++k) {
            const float d = sV[tid * KNB + k] - mu;
            ss += d * d;
        }
        sMu[tid]  = mu;
        sSig[tid] = fmaxf(sqrtf(ss * (1.0f / (KNB - 1))), 1e-3f);  // ddof=1, clipped
    }
    __syncthreads();

    if (tid < MTOK) {
        const int q = tid >> 5;
        const float g0 = __expf(log_gammas[0]);
        const float g1 = __expf(log_gammas[1]);
        const float g2 = __expf(log_gammas[2]);
        h16* t = sTok + tid * ST_TOK;
        t[0] = (h16)((cx - qx) * g0);
        t[1] = (h16)((cy - qy) * g1);
        t[2] = (h16)((cz - qz) * g2);
        t[3] = (h16)((val - sMu[q]) / sSig[q]);
    }

    // drain this wave's async weight copies, then rendezvous
    asm volatile("s_wait_asynccnt 0x0" ::: "memory");
    __syncthreads();

    // ---- GEMM1: tokens[128,4(pad32)] x w_in^T[32,128] -> sH, GELU ---------
    {
        const v16h a = ld_frag_a_lds(sTok, ST_TOK, wv * 16, 0);
#pragma unroll
        for (int nt = 0; nt < 8; ++nt) {
            v8f c = {};
            c = wmma_f16(a, ld_frag_b_lds(sWin, ST_TOK, nt * 16, 0), c);
            store_tile_gelu(sH, ST_H, wv * 16, nt * 16, c, b_in);
        }
    }
    __syncthreads();

    // ---- LayerNorm over D=128 (2 threads per row, shfl pair-reduce) -------
    {
        const int row = tid >> 1, part = tid & 1;
        h16* r = sH + row * ST_H + part * 64;
        float s = 0.f, ss = 0.f;
        for (int i = 0; i < 64; ++i) {
            const float x = (float)r[i];
            s += x; ss += x * x;
        }
        s  += __shfl_xor(s, 1, 32);
        ss += __shfl_xor(ss, 1, 32);
        const float m   = s * (1.0f / DD);
        const float var = ss * (1.0f / DD) - m * m;
        const float inv = rsqrtf(var + 1e-5f);
        for (int i = 0; i < 64; ++i) {
            const int   ch = part * 64 + i;
            const float x  = (float)r[i];
            r[i] = (h16)(((x - m) * inv) * ln1_g[ch] + ln1_b[ch]);
        }
    }
    __syncthreads();

    // ---- GEMM2: x[128,128] x w1^T[128,256] -> sX, GELU --------------------
    {
        const v16h a0 = ld_frag_a_lds(sH, ST_H, wv * 16, 0);
        const v16h a1 = ld_frag_a_lds(sH, ST_H, wv * 16, 32);
        const v16h a2 = ld_frag_a_lds(sH, ST_H, wv * 16, 64);
        const v16h a3 = ld_frag_a_lds(sH, ST_H, wv * 16, 96);
#pragma unroll
        for (int nt = 0; nt < 16; ++nt) {
            v8f c = {};
            c = wmma_f16(a0, ld_frag_b_lds(sW1, ST_H, nt * 16, 0),  c);
            c = wmma_f16(a1, ld_frag_b_lds(sW1, ST_H, nt * 16, 32), c);
            c = wmma_f16(a2, ld_frag_b_lds(sW1, ST_H, nt * 16, 64), c);
            c = wmma_f16(a3, ld_frag_b_lds(sW1, ST_H, nt * 16, 96), c);
            store_tile_gelu(sX, ST_X, wv * 16, nt * 16, c, b1);
        }
    }
    __syncthreads();

    // ---- GEMM3: x[128,256] x w2^T[256,128] -> sH (reuse), GELU ------------
    {
        v16h a[8];
#pragma unroll
        for (int kc = 0; kc < 8; ++kc)
            a[kc] = ld_frag_a_lds(sX, ST_X, wv * 16, kc * 32);
#pragma unroll
        for (int nt = 0; nt < 8; ++nt) {
            v8f c = {};
#pragma unroll
            for (int kc = 0; kc < 8; ++kc)
                c = wmma_f16(a[kc], ld_frag_b_lds(sW2, ST_X, nt * 16, kc * 32), c);
            store_tile_gelu(sH, ST_H, wv * 16, nt * 16, c, b2);
        }
    }
    __syncthreads();

    // ---- pool over K=32 neighbors: mean | max -> sPool[4][256] ------------
    for (int i = tid; i < DQ * DD; i += 256) {
        const int q = i >> 7, ch = i & 127;
        const h16* p = sH + (q * KNB) * ST_H + ch;
        float s = 0.f, mx = -3.4e38f;
        for (int k = 0; k < KNB; ++k) {
            const float x = (float)p[k * ST_H];
            s += x; mx = fmaxf(mx, x);
        }
        sPool[q * 256 + ch]       = s * (1.0f / KNB);
        sPool[q * 256 + 128 + ch] = mx;
    }
    __syncthreads();

    // ---- head LayerNorm over 256 (wave q handles query q) -----------------
    if (wv < DQ) {
        const float* row = sPool + wv * 256;
        float s = 0.f, ss = 0.f;
        for (int i = lane; i < 256; i += 32) {
            const float x = row[i];
            s += x; ss += x * x;
        }
#pragma unroll
        for (int m = 16; m >= 1; m >>= 1) {
            s  += __shfl_xor(s,  m, 32);
            ss += __shfl_xor(ss, m, 32);
        }
        const float mean = s * (1.0f / 256.0f);
        const float var  = ss * (1.0f / 256.0f) - mean * mean;
        const float inv  = rsqrtf(var + 1e-5f);
        for (int i = lane; i < 256; i += 32)
            sPln[wv * 256 + i] = (row[i] - mean) * inv * hln_g[i] + hln_b[i];
    }
    __syncthreads();

    // ---- head Lin(256->256)+GELU: thread = out channel, 4 queries ---------
    {
        const int ch = tid;
        const float* wrow = hw1 + ch * 256;
        float a0 = 0.f, a1 = 0.f, a2 = 0.f, a3 = 0.f;
        for (int k = 0; k < 256; ++k) {
            const float w = wrow[k];
            a0 += sPln[0 * 256 + k] * w;
            a1 += sPln[1 * 256 + k] * w;
            a2 += sPln[2 * 256 + k] * w;
            a3 += sPln[3 * 256 + k] * w;
        }
        const float bb = hb1[ch];
        sPv[0 * 256 + ch] = gelu_erf(a0 + bb);
        sPv[1 * 256 + ch] = gelu_erf(a1 + bb);
        sPv[2 * 256 + ch] = gelu_erf(a2 + bb);
        sPv[3 * 256 + ch] = gelu_erf(a3 + bb);
    }
    __syncthreads();

    // ---- head Lin(256->1), de-normalize -----------------------------------
    if (tid < DQ) {
        float u = hb2[0];
        for (int k = 0; k < 256; ++k) u += sPv[tid * 256 + k] * hw2[k];
        out[qblk + tid] = u * sSig[tid] + sMu[tid];
    }
}

extern "C" void kernel_launch(void* const* d_in, const int* in_sizes, int n_in,
                              void* d_out, int out_size, void* d_ws, size_t ws_size,
                              hipStream_t stream) {
    const float* xyt_q      = (const float*)d_in[0];
    const float* obs_coords = (const float*)d_in[1];
    const float* obs_vals   = (const float*)d_in[2];
    const int*   nb_idx     = (const int*)  d_in[3];
    const float* log_gammas = (const float*)d_in[4];
    const float* w_in       = (const float*)d_in[5];
    const float* b_in       = (const float*)d_in[6];
    const float* ln1_g      = (const float*)d_in[7];
    const float* ln1_b      = (const float*)d_in[8];
    const float* w1         = (const float*)d_in[9];
    const float* b1         = (const float*)d_in[10];
    const float* w2         = (const float*)d_in[11];
    const float* b2         = (const float*)d_in[12];
    const float* hln_g      = (const float*)d_in[13];
    const float* hln_b      = (const float*)d_in[14];
    const float* hw1        = (const float*)d_in[15];
    const float* hb1        = (const float*)d_in[16];
    const float* hw2        = (const float*)d_in[17];
    const float* hb2        = (const float*)d_in[18];

    h16* w_in_h = (h16*)d_ws;             // 128*32 halves (K padded)
    h16* w1_h   = w_in_h + DD * 32;       // 256*128 halves
    h16* w2_h   = w1_h   + FF * DD;       // 128*256 halves

    const int prep_elems = DD * 32 + FF * DD + DD * FF;
    ff_prep_weights<<<(prep_elems + 255) / 256, 256, 0, stream>>>(
        w_in, w1, w2, w_in_h, w1_h, w2_h);

    const size_t lds_bytes =
        (size_t)(DD * ST_TOK + FF * ST_H + DD * ST_X +          // staged weights
                 MTOK * ST_TOK + MTOK * ST_H + MTOK * ST_X)      // activations
            * sizeof(h16) +
        (size_t)(DQ * KNB + DQ + DQ + 3 * DQ * 256) * sizeof(float);

    (void)hipFuncSetAttribute((const void*)ff_main,
                              hipFuncAttributeMaxDynamicSharedMemorySize,
                              (int)lds_bytes);

    const int Q = in_sizes[0] / 3;   // xyt_q is [Q,3]
    ff_main<<<Q / DQ, 256, lds_bytes, stream>>>(
        xyt_q, obs_coords, obs_vals, nb_idx, log_gammas,
        b_in, ln1_g, ln1_b, b1, b2, hln_g, hln_b, hw1, hb1, hw2, hb2,
        w_in_h, w1_h, w2_h, (float*)d_out);
}
